// modelGCN_Hetero_47760036331532
// MI455X (gfx1250) — matrix operand order repaired and checked by reading, not compiled, not run on hardware
//
#include <hip/hip_runtime.h>
#include <math.h>

// ---- CDNA5 WMMA types (wave32; 16x16 f32 D = 8 VGPRs, bf16 A/B frag = 16 elems/lane)
typedef __attribute__((ext_vector_type(16))) __bf16 v16bf;
typedef __attribute__((ext_vector_type(8)))  __bf16 v8bf;
typedef __attribute__((ext_vector_type(8)))  float  v8f;

#define NNODES 50000
#define NEDGES 800000

// ---------------- elementwise / graph-prep kernels ----------------

__global__ void k_zero(float* __restrict__ p, long long n) {
  long long i = (long long)blockIdx.x * blockDim.x + threadIdx.x;
  if (i < n) p[i] = 0.0f;
}

__global__ void k_deg(const int* __restrict__ ei, float* __restrict__ deg, int E) {
  int e = blockIdx.x * blockDim.x + threadIdx.x;
  if (e < E) atomicAdd(&deg[ei[E + e]], 1.0f);   // degree over destination
}

__global__ void k_dinv(float* __restrict__ d, int n) {
  int i = blockIdx.x * blockDim.x + threadIdx.x;
  if (i < n) { float v = d[i]; d[i] = (v > 0.0f) ? rsqrtf(v) : 0.0f; }
}

__global__ void k_norm(const int* __restrict__ ei, const float* __restrict__ dinv,
                       float* __restrict__ norm, int E) {
  int e = blockIdx.x * blockDim.x + threadIdx.x;
  if (e < E) norm[e] = dinv[ei[e]] * dinv[ei[E + e]];
}

__global__ void k_cvt_bf16(const float* __restrict__ src, __bf16* __restrict__ dst,
                           long long n) {
  long long i = (long long)blockIdx.x * blockDim.x + threadIdx.x;
  if (i < n) dst[i] = (__bf16)src[i];
}

// ---------------- one-time weight pack into WMMA B-fragment layout ----------------
// Packed element ((ct*kch + kc32)*32 + lane)*16 + j  ==  B[k][col] with
//   k = kc32*32 + 8*(lane>>4) + (j&7) + 16*(j>>3), col = ct*16 + (lane&15).
// transposed==0: W is [K,M] row-major (B[k][n] = W[k*M+n]), M == Mvalid.
// transposed==1: W is [Mvalid,K] row-major torch layout (B[k][n] = W[n*K+k]).
// Cols >= Mvalid are zero-padded, so the GEMM needs no guards.
__global__ void k_pack_b(const float* __restrict__ W, __bf16* __restrict__ P,
                         int K, int M, int Mvalid, int transposed) {
  int t = blockIdx.x * blockDim.x + threadIdx.x;   // t = (ct*kch + kc32)*32 + lane
  int colTiles = (Mvalid + 15) >> 4;
  int kch = K >> 5;
  if (t >= colTiles * kch * 32) return;
  int lane = t & 31;
  int kc32 = (t >> 5) % kch;
  int ct   = (t >> 5) / kch;
  int h = lane >> 4, n = lane & 15;
  int col = ct * 16 + n;
  __bf16* out = P + (size_t)t * 16;
#pragma unroll
  for (int j = 0; j < 16; ++j) {
    int k = kc32 * 32 + 8 * h + (j & 7) + 16 * (j >> 3);
    float v = 0.0f;
    if (col < Mvalid)
      v = transposed ? W[(size_t)col * K + k] : W[(size_t)k * M + col];
    out[j] = (__bf16)v;
  }
}

// ---------------- WMMA GEMM: C[n,m] = sum_k A[n,k]*B[k,m] (+ bias) ----------------
// A: bf16 row-major [nRows,K]; Bp: packed bf16 fragments (see k_pack_b).
// One wave -> one 16x16 tile. Inner loop: 2x b128 (A) + 32B (B) + 1 wmma per K=32.
__global__ void k_gemm_bf16(const __bf16* __restrict__ A, const __bf16* __restrict__ Bp,
                            const float* __restrict__ bias, float* __restrict__ C,
                            int nRows, int K, int Mvalid, int ldC) {
  int wave = (int)(((long long)blockIdx.x * blockDim.x + threadIdx.x) >> 5);
  int lane = threadIdx.x & 31;
  int colTiles = (Mvalid + 15) >> 4;
  int rowTiles = nRows >> 4;
  if (wave >= colTiles * rowTiles) return;     // wave-uniform branch
  int rt = wave / colTiles, ct = wave - rt * colTiles;
  int row0 = rt << 4, col0 = ct << 4;
  int h = lane >> 4, m = lane & 15;
  int kch = K >> 5;
  const __bf16* arow = A + (size_t)(row0 + m) * K + 8 * h;
  const v16bf* bp = (const v16bf*)Bp + (size_t)ct * kch * 32 + lane;
  v8f acc = {};
  for (int kc = 0; kc < kch; ++kc) {
    v8bf alo = *(const v8bf*)(arow + (size_t)kc * 32);        // K = kc*32+8h+0..7
    v8bf ahi = *(const v8bf*)(arow + (size_t)kc * 32 + 16);   // K = kc*32+8h+16..23
    v16bf a = __builtin_shufflevector(alo, ahi, 0, 1, 2, 3, 4, 5, 6, 7,
                                      8, 9, 10, 11, 12, 13, 14, 15);
    v16bf b = bp[(size_t)kc * 32];                            // pre-swizzled fragment
    acc = __builtin_amdgcn_wmma_f32_16x16x32_bf16(false, a, false, b,
                                                  (short)0, acc, false, false);
  }
  int col = col0 + m;
  if (col < Mvalid) {
    float bv = (bias != nullptr) ? bias[col] : 0.0f;
#pragma unroll
    for (int r = 0; r < 8; ++r)
      C[(size_t)(row0 + r + 8 * h) * ldC + col] = acc[r] + bv;  // D: VGPR r -> row r+8h
  }
}

// ---------------- edge scatter-add: acc[dst] += norm * T[src] (D=128) ----------------
// 32 threads/edge, float4 each -> coalesced gather + f32 atomics (all L2-resident).
__global__ void k_scatter(const int* __restrict__ ei, const float* __restrict__ norm,
                          const float* __restrict__ T, float* __restrict__ acc, int E) {
  long long t = (long long)blockIdx.x * blockDim.x + threadIdx.x;
  long long e = t >> 5;
  if (e >= E) return;
  int c = ((int)t & 31) << 2;
  int s = ei[e], d = ei[(size_t)E + e];
  float w = norm[e];
  const float4 v = *(const float4*)(T + (size_t)s * 128 + c);
  float* o = acc + (size_t)d * 128 + c;
  atomicAdd(o + 0, v.x * w);
  atomicAdd(o + 1, v.y * w);
  atomicAdd(o + 2, v.z * w);
  atomicAdd(o + 3, v.w * w);
}

// fused bias + ReLU, emitting bf16 activations for the next WMMA GEMM
__global__ void k_bias_relu128_bf16(const float* __restrict__ acc,
                                    const float* __restrict__ b,
                                    __bf16* __restrict__ out, long long n) {
  long long i = (long long)blockIdx.x * blockDim.x + threadIdx.x;
  if (i < n) {
    float v = acc[i] + b[(int)(i & 127)];
    out[i] = (__bf16)(v > 0.0f ? v : 0.0f);
  }
}

__global__ void k_softmax(float* __restrict__ x, int nRows, int D) {
  int r = blockIdx.x * blockDim.x + threadIdx.x;
  if (r >= nRows) return;
  float* p = x + (size_t)r * D;
  float mx = -INFINITY;
  for (int j = 0; j < D; ++j) mx = fmaxf(mx, p[j]);
  float s = 0.0f;
  for (int j = 0; j < D; ++j) { float e = expf(p[j] - mx); p[j] = e; s += e; }
  float inv = 1.0f / s;
  for (int j = 0; j < D; ++j) p[j] *= inv;
}

// ---------------- launch ----------------

extern "C" void kernel_launch(void* const* d_in, const int* in_sizes, int n_in,
                              void* d_out, int out_size, void* d_ws, size_t ws_size,
                              hipStream_t stream) {
  const float* x   = (const float*)d_in[0];
  const int*   ei  = (const int*)  d_in[1];
  const float* W0  = (const float*)d_in[2];
  const float* b0  = (const float*)d_in[3];
  const float* W1  = (const float*)d_in[4];
  const float* b1  = (const float*)d_in[5];
  const float* W2  = (const float*)d_in[6];
  const float* b2  = (const float*)d_in[7];
  const float* L0W = (const float*)d_in[8];
  const float* L0b = (const float*)d_in[9];
  const float* L1W = (const float*)d_in[10];
  const float* L1b = (const float*)d_in[11];
  float* out = (float*)d_out;
  float* ws  = (float*)d_ws;

  const int N = NNODES, E = NEDGES;
  // workspace layout (float offsets, 32B-aligned): ~80.4 MB total
  float*  dinv = ws;                           // N (deg -> rsqrt in place)
  float*  nrm  = ws + 50048;                   // E
  float*  bufA = ws + 850048;                  // N*128 f32 (GEMM out / logits)
  float*  bufB = ws + 7250048;                 // N*128 f32 (scatter accumulator)
  __bf16* hbf  = (__bf16*)(ws + 13650048);     // bf16 activations (up to N*256)
  __bf16* wpk  = (__bf16*)(ws + 20050048);     // packed bf16 weights
  __bf16* W0p = wpk;                           // 8 tiles * 8 chunks * 512 = 32768
  __bf16* W1p = wpk + 32768;                   // 8*4*512 = 16384
  __bf16* W2p = wpk + 49152;                   // 16384
  __bf16* L0p = wpk + 65536;                   // 4*4*512 = 8192
  __bf16* L1p = wpk + 73728;                   // 3*2*512 = 3072

  const int B = 256;
  // graph norm
  k_zero<<<(N + B - 1) / B, B, 0, stream>>>(dinv, (long long)N);
  k_deg <<<(E + B - 1) / B, B, 0, stream>>>(ei, dinv, E);
  k_dinv<<<(N + B - 1) / B, B, 0, stream>>>(dinv, N);
  k_norm<<<(E + B - 1) / B, B, 0, stream>>>(ei, dinv, nrm, E);

  // one-time weight packs (tiny)
  k_pack_b<<<(2048 + B - 1) / B, B, 0, stream>>>(W0,  W0p, 256, 128, 128, 0);
  k_pack_b<<<(1024 + B - 1) / B, B, 0, stream>>>(W1,  W1p, 128, 128, 128, 0);
  k_pack_b<<<(1024 + B - 1) / B, B, 0, stream>>>(W2,  W2p, 128, 128, 128, 0);
  k_pack_b<<<( 512 + B - 1) / B, B, 0, stream>>>(L0W, L0p, 128, 0,   64,  1);
  k_pack_b<<<( 192 + B - 1) / B, B, 0, stream>>>(L1W, L1p, 64,  0,   40,  1);

  // x -> bf16 once
  const long long nx = (long long)N * 256;
  k_cvt_bf16<<<(int)((nx + B - 1) / B), B, 0, stream>>>(x, hbf, nx);

  const long long nd = (long long)N * 128;
  const int convTiles  = (N / 16) * (128 / 16);          // 25000 waves
  const int convBlocks = (convTiles + 3) / 4;            // 4 waves / 128-thread block
  const int ndB   = (int)((nd + B - 1) / B);
  const int scatB = (int)(((long long)E * 32 + B - 1) / B);

  // ---- conv0: x[N,256] -> h1
  k_gemm_bf16<<<convBlocks, 128, 0, stream>>>(hbf, W0p, nullptr, bufA, N, 256, 128, 128);
  k_zero<<<ndB, B, 0, stream>>>(bufB, nd);
  k_scatter<<<scatB, B, 0, stream>>>(ei, nrm, bufA, bufB, E);
  k_bias_relu128_bf16<<<ndB, B, 0, stream>>>(bufB, b0, hbf, nd);   // h1 (bf16)

  // ---- conv1
  k_gemm_bf16<<<convBlocks, 128, 0, stream>>>(hbf, W1p, nullptr, bufA, N, 128, 128, 128);
  k_zero<<<ndB, B, 0, stream>>>(bufB, nd);
  k_scatter<<<scatB, B, 0, stream>>>(ei, nrm, bufA, bufB, E);
  k_bias_relu128_bf16<<<ndB, B, 0, stream>>>(bufB, b1, hbf, nd);   // h2

  // ---- conv2
  k_gemm_bf16<<<convBlocks, 128, 0, stream>>>(hbf, W2p, nullptr, bufA, N, 128, 128, 128);
  k_zero<<<ndB, B, 0, stream>>>(bufB, nd);
  k_scatter<<<scatB, B, 0, stream>>>(ei, nrm, bufA, bufB, E);
  k_bias_relu128_bf16<<<ndB, B, 0, stream>>>(bufB, b2, hbf, nd);   // h3

  // ---- head: softmax(h3 @ L0_W.T + L0_b) -> softmax(s0 @ L1_W.T + L1_b)
  {
    int tiles = (N / 16) * (64 / 16);
    k_gemm_bf16<<<(tiles + 3) / 4, 128, 0, stream>>>(hbf, L0p, L0b, bufA, N, 128, 64, 64);
  }
  k_softmax<<<(N + B - 1) / B, B, 0, stream>>>(bufA, N, 64);
  {
    const long long ns = (long long)N * 64;
    k_cvt_bf16<<<(int)((ns + B - 1) / B), B, 0, stream>>>(bufA, hbf, ns);  // s0 -> bf16
  }
  {
    int tiles = (N / 16) * 3;   // 40 classes -> 3 col tiles (B zero-padded at pack time)
    k_gemm_bf16<<<(tiles + 3) / 4, 128, 0, stream>>>(hbf, L1p, L1b, out, N, 64, 40, 40);
  }
  k_softmax<<<(N + B - 1) / B, B, 0, stream>>>(out, N, 40);
}
